// AttentionPooling_15994458210502
// MI455X (gfx1250) — matrix operand order repaired and checked by reading, not compiled, not run on hardware
//
#include <hip/hip_runtime.h>
#include <hip/hip_bf16.h>
#include <stdint.h>

// ---------------- problem constants (match reference) ----------------
#define G_NUM   2048      // NUM_GRAPHS
#define H_DIM   256       // H
#define H_HALF  128       // H/2
#define LDS_K_PAD 264     // 256 + 8 ushorts pad: column stride = 132 dwords -> no bank conflicts

typedef __attribute__((ext_vector_type(16))) __bf16          v16bf;
typedef __attribute__((ext_vector_type(16))) unsigned short  v16us;
typedef __attribute__((ext_vector_type(8)))  unsigned short  v8us;
typedef __attribute__((ext_vector_type(8)))  float           v8f;
typedef __attribute__((ext_vector_type(4)))  float           v4f;

// native f32 -> bf16 convert (backend selects v_cvt*_bf16_f32 on gfx1250)
__device__ __forceinline__ unsigned short f2bf_bits(float f) {
  return __builtin_bit_cast(unsigned short, (__bf16)f);
}

// Pack 16 raw floats (two 8-float runs) into an A fragment, wave32 layout:
//  lane = {m = lane&15, kh = lane>>4}; elems 0..7 -> K = kh*8+e, elems 8..15 -> K = 16+kh*8+(e-8)
__device__ __forceinline__ v16bf cvt_a_frag(v4f a0, v4f a1, v4f b0, v4f b1) {
  v16bf t;
#pragma unroll
  for (int i = 0; i < 4; ++i) {
    t[i]      = (__bf16)a0[i];
    t[4 + i]  = (__bf16)a1[i];
    t[8 + i]  = (__bf16)b0[i];
    t[12 + i] = (__bf16)b1[i];
  }
  return t;
}

// B fragment (32x16 bf16, KxN) from LDS (column-major, padded stride):
//  lane = {n = lane&15, kh = lane>>4}; elems 0..7 -> K = k0 + kh*8 + e, elems 8..15 -> K = k0 + 16 + kh*8 + (e-8), column n
__device__ __forceinline__ v16bf load_b_frag(const unsigned short* __restrict__ lds,
                                             int col, int k0, int kh) {
  const v8us* p = (const v8us*)(lds + col * LDS_K_PAD + k0 + kh * 8); // 16B aligned
  v8us lo = p[0], hi = p[2];
  v16us t;
#pragma unroll
  for (int i = 0; i < 8; ++i) { t[i] = lo[i]; t[8 + i] = hi[i]; }
  return __builtin_bit_cast(v16bf, t);
}

// -------------------------------------------------------------------------
// Kernel 0: zero graph_emb region of d_out
// -------------------------------------------------------------------------
extern "C" __global__ void ap_zero(float* __restrict__ p, int n) {
  int i = blockIdx.x * blockDim.x + threadIdx.x;
  if (i < n) p[i] = 0.0f;
}

// -------------------------------------------------------------------------
// Kernel 1: scores[i] = relu(x_i @ gate_w1 + b1) @ gate_w2 + b2
// 8 waves/block; each wave owns one 16-row tile; gate_w1 bf16 in LDS.
// -------------------------------------------------------------------------
extern "C" __global__ __launch_bounds__(256)
void ap_gate_scores(const float* __restrict__ x,  const float* __restrict__ gw1,
                    const float* __restrict__ gb1, const float* __restrict__ gw2,
                    const float* __restrict__ gb2, float* __restrict__ scores, int N)
{
  extern __shared__ unsigned short lds[];                 // H_HALF cols * LDS_K_PAD
  const int tid = threadIdx.x;
  for (int idx = tid; idx < H_HALF * H_DIM; idx += 256) { // gw1 is [K=256][col=128] row-major
    int col = idx >> 8, k = idx & 255;
    lds[col * LDS_K_PAD + k] = f2bf_bits(gw1[k * H_HALF + col]);
  }
  __syncthreads();

  const int wave = tid >> 5, lane = tid & 31;
  const int n = lane & 15, kh = lane >> 4;
  const int ntiles = (N + 15) >> 4;
  const float b2 = gb2[0];

  for (int tile = blockIdx.x * 8 + wave; tile < ntiles; tile += gridDim.x * 8) {
    const int row0 = tile << 4;
    int arow = row0 + n; if (arow >= N) arow = N - 1;     // clamp: garbage rows masked at write
    const float* xrow = x + (size_t)arow * H_DIM;

    v8f acc[8] = {};
    // software-pipelined A stream: raw floats for step k0 live in r0..r3
    v4f r0 = *(const v4f*)(xrow + kh * 8);
    v4f r1 = *(const v4f*)(xrow + kh * 8 + 4);
    v4f r2 = *(const v4f*)(xrow + 16 + kh * 8);
    v4f r3 = *(const v4f*)(xrow + 16 + kh * 8 + 4);
#pragma unroll 1
    for (int k0 = 0; k0 < H_DIM; k0 += 32) {
      v16bf A = cvt_a_frag(r0, r1, r2, r3);
      if (k0 + 32 < H_DIM) {                              // prefetch next K-step
        const float* p = xrow + (k0 + 32) + kh * 8;
        r0 = *(const v4f*)(p);
        r1 = *(const v4f*)(p + 4);
        r2 = *(const v4f*)(p + 16);
        r3 = *(const v4f*)(p + 20);
      }
#pragma unroll
      for (int ct = 0; ct < 8; ++ct) {
        v16bf B = load_b_frag(lds, ct * 16 + n, k0, kh);
        acc[ct] = __builtin_amdgcn_wmma_f32_16x16x32_bf16(
            false, A, false, B, (short)0, acc[ct], false, false);
      }
    }
    // Epilogue: D[v] holds row m = kh*8+v, column n+16*ct. Fuse bias, ReLU, dot(w2).
    float sv[8];
#pragma unroll
    for (int v = 0; v < 8; ++v) sv[v] = 0.0f;
#pragma unroll
    for (int ct = 0; ct < 8; ++ct) {
      const int col = ct * 16 + n;
      const float b1c = gb1[col], w2c = gw2[col];
#pragma unroll
      for (int v = 0; v < 8; ++v) {
        float h = acc[ct][v] + b1c;
        h = h > 0.0f ? h : 0.0f;
        sv[v] = fmaf(h, w2c, sv[v]);
      }
    }
    // reduce over the 16 columns (lanes within each half)
#pragma unroll
    for (int off = 1; off <= 8; off <<= 1)
#pragma unroll
      for (int v = 0; v < 8; ++v) sv[v] += __shfl_xor(sv[v], off, 32);
    if (n == 0) {
#pragma unroll
      for (int v = 0; v < 8; ++v) {
        int row = row0 + kh * 8 + v;
        if (row < N) scores[row] = sv[v] + b2;
      }
    }
  }
}

// -------------------------------------------------------------------------
// Kernel 2: per-graph stable softmax. batch is sorted -> one wave per graph,
// binary search the segment, shuffle-reduce max and sum(exp). Deterministic.
// -------------------------------------------------------------------------
extern "C" __global__ __launch_bounds__(32)
void ap_segment_softmax(const float* __restrict__ scores, const int* __restrict__ batch,
                        float* __restrict__ alpha, int N)
{
  const int g = blockIdx.x;
  const int lane = threadIdx.x;

  int lo = 0, hi = N;                                   // lower_bound(g)
  while (lo < hi) { int mid = (lo + hi) >> 1; if (batch[mid] < g) lo = mid + 1; else hi = mid; }
  const int s0 = lo;
  hi = N;                                               // lower_bound(g+1)
  while (lo < hi) { int mid = (lo + hi) >> 1; if (batch[mid] < g + 1) lo = mid + 1; else hi = mid; }
  const int s1 = lo;

  float m = -3.402823466e38f;
  for (int i = s0 + lane; i < s1; i += 32) m = fmaxf(m, scores[i]);
#pragma unroll
  for (int off = 1; off <= 16; off <<= 1) m = fmaxf(m, __shfl_xor(m, off, 32));
  if (s1 <= s0) m = 0.0f;                               // empty-segment guard (matches reference)

  float s = 0.0f;
  for (int i = s0 + lane; i < s1; i += 32) s += __expf(scores[i] - m);
#pragma unroll
  for (int off = 1; off <= 16; off <<= 1) s += __shfl_xor(s, off, 32);

  const float inv = (s > 0.0f) ? (1.0f / s) : 0.0f;
  for (int i = s0 + lane; i < s1; i += 32) alpha[i] = __expf(scores[i] - m) * inv;
}

// -------------------------------------------------------------------------
// Kernel 3: graph_emb += alpha_i * relu(x_i @ feat_w + fb), segment-summed.
// Wave pairs split the 256 columns (8 C-tiles each). Sorted-batch fast path:
// single-graph tiles pre-reduce 16 rows in-register -> 16 atomics/tile.
// -------------------------------------------------------------------------
extern "C" __global__ __launch_bounds__(256)
void ap_feat_pool(const float* __restrict__ x, const float* __restrict__ fw,
                  const float* __restrict__ fb, const int* __restrict__ batch,
                  const float* __restrict__ alpha, float* __restrict__ gemb, int N)
{
  extern __shared__ unsigned short lds[];                 // H_DIM cols * LDS_K_PAD
  const int tid = threadIdx.x;
  for (int idx = tid; idx < H_DIM * H_DIM; idx += 256) {  // fw is [K=256][col=256] row-major
    int col = idx >> 8, k = idx & 255;
    lds[col * LDS_K_PAD + k] = f2bf_bits(fw[k * H_DIM + col]);
  }
  __syncthreads();

  const int wave = tid >> 5, lane = tid & 31;
  const int n = lane & 15, kh = lane >> 4;
  const int ntiles = (N + 15) >> 4;
  const int nwork = ntiles * 2;

  for (int u = blockIdx.x * 8 + wave; u < nwork; u += gridDim.x * 8) {
    const int tile = u >> 1;
    const int coff = (u & 1) * 128;                       // which half of the 256 columns
    const int row0 = tile << 4;
    int arow = row0 + n; if (arow >= N) arow = N - 1;
    const float* xrow = x + (size_t)arow * H_DIM;

    v8f acc[8] = {};
    v4f r0 = *(const v4f*)(xrow + kh * 8);
    v4f r1 = *(const v4f*)(xrow + kh * 8 + 4);
    v4f r2 = *(const v4f*)(xrow + 16 + kh * 8);
    v4f r3 = *(const v4f*)(xrow + 16 + kh * 8 + 4);
#pragma unroll 1
    for (int k0 = 0; k0 < H_DIM; k0 += 32) {
      v16bf A = cvt_a_frag(r0, r1, r2, r3);
      if (k0 + 32 < H_DIM) {                              // prefetch next K-step
        const float* p = xrow + (k0 + 32) + kh * 8;
        r0 = *(const v4f*)(p);
        r1 = *(const v4f*)(p + 4);
        r2 = *(const v4f*)(p + 16);
        r3 = *(const v4f*)(p + 20);
      }
#pragma unroll
      for (int ct = 0; ct < 8; ++ct) {
        v16bf B = load_b_frag(lds, coff + ct * 16 + n, k0, kh);
        acc[ct] = __builtin_amdgcn_wmma_f32_16x16x32_bf16(
            false, A, false, B, (short)0, acc[ct], false, false);
      }
    }

    // per-lane alpha / graph id for its 8 rows (m = kh*8 + v)
    float av[8]; int bg[8];
#pragma unroll
    for (int v = 0; v < 8; ++v) {
      int row = row0 + kh * 8 + v;
      bool ok = row < N;
      av[v] = ok ? alpha[row] : 0.0f;
      bg[v] = ok ? batch[row] : 0;
    }

    const bool fulltile = (row0 + 15) < N;
    const int g0 = batch[row0];
    const bool single = fulltile && (batch[row0 + 15] == g0);  // sorted => whole tile same graph

    if (single) {
#pragma unroll
      for (int ct = 0; ct < 8; ++ct) {
        const int col = coff + ct * 16 + n;
        const float bc = fb[col];
        float s = 0.0f;
#pragma unroll
        for (int v = 0; v < 8; ++v) {
          float h = acc[ct][v] + bc;
          h = h > 0.0f ? h : 0.0f;
          s = fmaf(av[v], h, s);
        }
        s += __shfl_xor(s, 16, 32);                       // combine row halves 0-7 / 8-15
        if (kh == 0) atomicAdd(&gemb[(size_t)g0 * H_DIM + col], s);
      }
    } else {
#pragma unroll
      for (int ct = 0; ct < 8; ++ct) {
        const int col = coff + ct * 16 + n;
        const float bc = fb[col];
#pragma unroll
        for (int v = 0; v < 8; ++v) {
          int row = row0 + kh * 8 + v;
          if (row < N) {
            float h = acc[ct][v] + bc;
            h = h > 0.0f ? h : 0.0f;
            atomicAdd(&gemb[(size_t)bg[v] * H_DIM + col], av[v] * h);
          }
        }
      }
    }
  }
}

// -------------------------------------------------------------------------
extern "C" void kernel_launch(void* const* d_in, const int* in_sizes, int n_in,
                              void* d_out, int out_size, void* d_ws, size_t ws_size,
                              hipStream_t stream) {
  const float* x     = (const float*)d_in[0];
  const int*   batch = (const int*)d_in[1];
  const float* gw1   = (const float*)d_in[2];
  const float* gb1   = (const float*)d_in[3];
  const float* gw2   = (const float*)d_in[4];
  const float* gb2   = (const float*)d_in[5];
  const float* fw    = (const float*)d_in[6];
  const float* fb    = (const float*)d_in[7];
  const int N = in_sizes[0] / H_DIM;

  float* gemb   = (float*)d_out;                         // [G_NUM, H_DIM]
  float* alpha  = gemb + (size_t)G_NUM * H_DIM;          // [N] (second output)
  float* scores = (float*)d_ws;                          // [N] scratch

  const int lds1 = H_HALF * LDS_K_PAD * (int)sizeof(unsigned short); //  67.6 KB
  const int lds3 = H_DIM  * LDS_K_PAD * (int)sizeof(unsigned short); // 135.2 KB
  (void)hipFuncSetAttribute((const void*)ap_gate_scores,
                            hipFuncAttributeMaxDynamicSharedMemorySize, lds1);
  (void)hipFuncSetAttribute((const void*)ap_feat_pool,
                            hipFuncAttributeMaxDynamicSharedMemorySize, lds3);

  ap_zero<<<(G_NUM * H_DIM + 255) / 256, 256, 0, stream>>>(gemb, G_NUM * H_DIM);

  const int ntiles  = (N + 15) / 16;
  const int blocks1 = (ntiles + 7) / 8;
  ap_gate_scores<<<blocks1, 256, lds1, stream>>>(x, gw1, gb1, gw2, gb2, scores, N);

  ap_segment_softmax<<<G_NUM, 32, 0, stream>>>(scores, batch, alpha, N);

  const int blocks3 = (ntiles * 2 + 7) / 8;
  ap_feat_pool<<<blocks3, 256, lds3, stream>>>(x, fw, fb, batch, alpha, gemb, N);

  (void)n_in; (void)out_size; (void)ws_size;
}
